// TransMIL_60224031425194
// MI455X (gfx1250) — matrix operand order repaired and checked by reading, not compiled, not run on hardware
//
#include <hip/hip_runtime.h>
#include <hip/hip_bf16.h>

typedef __attribute__((ext_vector_type(16))) _Float16 v16h;
typedef __attribute__((ext_vector_type(8)))  float    v8f;
typedef _Float16 f16;

#define DEV __device__ __forceinline__

// CDNA5 async global->LDS copy (16 bytes per lane), tracked by ASYNCcnt.
#define ASYNC_COPY16(ldsPtr, gPtr)                                         \
  asm volatile("global_load_async_to_lds_b128 %0, %1, off"                 \
               :: "v"((unsigned)(unsigned long long)(ldsPtr)),             \
                  "v"((unsigned long long)(gPtr)) : "memory")

DEV void wait_async0() { asm volatile("s_wait_asynccnt 0x0" ::: "memory"); }

// ======================= batched WMMA GEMM (f16 in, f32 acc) =======================
// C[z] = alpha * A[z] @ opB(B[z]) (+bias)(+relu).  M mult of 128, N mult of 64, K mult of 32.
// opB==0: B is KxN row-major.  opB==1: B is NxK row-major (logical B^T stored).
__global__ __launch_bounds__(256) void k_gemm(
    const f16* __restrict__ A, const f16* __restrict__ B, float* __restrict__ C,
    const float* __restrict__ bias,
    int M, int N, int K, long lda, long ldb, long ldc,
    long sA, long sB, long sC, int opB, int epi, float alpha)
{
  __shared__ __align__(16) f16 As[2][128][32];   // [buf][m][k]
  __shared__ __align__(16) f16 Bs[2][64][32];    // [buf][n][k]
  const int z = blockIdx.z;
  A += (long)z * sA; B += (long)z * sB; C += (long)z * sC;
  const int bm = blockIdx.y * 128;
  const int bn = blockIdx.x * 64;
  const int t    = threadIdx.x;
  const int lane = t & 31;
  const int wave = t >> 5;       // 0..7
  const int wm   = wave >> 1;    // 0..3 -> 32-row slabs
  const int wn   = wave & 1;     // 0..1 -> 32-col slabs
  const int half = lane >> 4;
  const int l16  = lane & 15;
  v8f acc00 = {}, acc01 = {}, acc10 = {}, acc11 = {};

  const int ar  = t >> 2,  ac  = (t & 3) << 3;   // A / B^T staging: 16B chunk per lane
  const int bk  = t >> 3,  bn8 = (t & 7) << 3;   // B (KxN) staging

  auto stage = [&](int k0, int buf) {
    // A tile: 128x32 halves = 512 x 16B chunks, 2 per thread (async, no VGPR round trip)
    ASYNC_COPY16(&As[buf][ar][ac],      A + (long)(bm + ar) * lda + k0 + ac);
    ASYNC_COPY16(&As[buf][ar + 64][ac], A + (long)(bm + ar + 64) * lda + k0 + ac);
    if (k0 + 64 < K) __builtin_prefetch(A + (long)(bm + ar) * lda + k0 + 64, 0, 0);
    if (opB == 0) {
      __align__(16) f16 tmp[8];
      *(uint4*)tmp = *(const uint4*)(B + (long)(k0 + bk) * ldb + bn + bn8);
      #pragma unroll
      for (int i = 0; i < 8; ++i) Bs[buf][bn8 + i][bk] = tmp[i];
    } else {
      ASYNC_COPY16(&Bs[buf][ar][ac], B + (long)(bn + ar) * ldb + k0 + ac);
    }
  };

  stage(0, 0);
  wait_async0();
  __syncthreads();
  int s = 0;
  for (int k0 = 0; k0 < K; k0 += 32) {
    if (k0 + 32 < K) stage(k0 + 32, s ^ 1);
    // A frags: row-major LDS; elems 0..7 -> K = 8*half+0..7 ; 8..15 -> K = 16+8*half+0..7
    union { v16h v; uint4 q[2]; } a0, a1, b0, b1;
    const f16* ap0 = &As[s][wm * 32 + l16][half * 8];
    a0.q[0] = *(const uint4*)ap0; a0.q[1] = *(const uint4*)(ap0 + 16);
    const f16* ap1 = &As[s][wm * 32 + 16 + l16][half * 8];
    a1.q[0] = *(const uint4*)ap1; a1.q[1] = *(const uint4*)(ap1 + 16);
    // B frags: n-major LDS; elems 0..7 -> K = 8*half+0..7 ; 8..15 -> K = 16+8*half+0..7
    const f16* bp0 = &Bs[s][wn * 32 + l16][half * 8];
    b0.q[0] = *(const uint4*)bp0; b0.q[1] = *(const uint4*)(bp0 + 16);
    const f16* bp1 = &Bs[s][wn * 32 + 16 + l16][half * 8];
    b1.q[0] = *(const uint4*)bp1; b1.q[1] = *(const uint4*)(bp1 + 16);
    acc00 = __builtin_amdgcn_wmma_f32_16x16x32_f16(false, a0.v, false, b0.v, (short)0, acc00, false, false);
    acc01 = __builtin_amdgcn_wmma_f32_16x16x32_f16(false, a0.v, false, b1.v, (short)0, acc01, false, false);
    acc10 = __builtin_amdgcn_wmma_f32_16x16x32_f16(false, a1.v, false, b0.v, (short)0, acc10, false, false);
    acc11 = __builtin_amdgcn_wmma_f32_16x16x32_f16(false, a1.v, false, b1.v, (short)0, acc11, false, false);
    wait_async0();
    __syncthreads();
    s ^= 1;
  }
  const int nc = bn + wn * 32 + l16;
  #pragma unroll
  for (int r = 0; r < 8; ++r) {
    int m0 = bm + wm * 32 + r + 8 * half;
    int m1 = m0 + 16;
    float v00 = acc00[r] * alpha, v01 = acc01[r] * alpha;
    float v10 = acc10[r] * alpha, v11 = acc11[r] * alpha;
    if (epi >= 1) {
      float bl = bias[nc], bh = bias[nc + 16];
      v00 += bl; v10 += bl; v01 += bh; v11 += bh;
    }
    if (epi == 2) {
      v00 = fmaxf(v00, 0.f); v01 = fmaxf(v01, 0.f);
      v10 = fmaxf(v10, 0.f); v11 = fmaxf(v11, 0.f);
    }
    C[(long)m0 * ldc + nc]      = v00;
    C[(long)m0 * ldc + nc + 16] = v01;
    C[(long)m1 * ldc + nc]      = v10;
    C[(long)m1 * ldc + nc + 16] = v11;
  }
}

// ======================= elementwise / reduction kernels =======================
__global__ __launch_bounds__(256) void k_cast_f16(const float* __restrict__ in, f16* __restrict__ out, long n) {
  long i = blockIdx.x * 256L + threadIdx.x;
  if (i < n) out[i] = (f16)in[i];
}

__global__ __launch_bounds__(256) void k_zero_pad(f16* __restrict__ xp) {
  long e = blockIdx.x * 256L + threadIdx.x;
  if (e >= 8L * 59 * 512) return;
  int b = (int)(e / 30208); int rem = (int)(e % 30208);
  xp[(long)b * 3145728 + rem] = (f16)0.f;
}

__global__ __launch_bounds__(256) void k_build_seq(const float* __restrict__ h0, const float* __restrict__ cls,
                                                   float* __restrict__ seq) {
  long e = blockIdx.x * 256L + threadIdx.x;
  if (e >= 8L * 6085 * 512) return;
  int d = (int)(e & 511);
  long rn = e >> 9;
  int n = (int)(rn % 6085);
  int b = (int)(rn / 6085);
  float val;
  if (n == 0) val = cls[d];
  else { int i = n - 1; if (i >= 6000) i -= 6000; val = h0[((long)b * 6000 + i) * 512 + d]; }
  seq[e] = val;
}

__global__ __launch_bounds__(256) void k_layernorm_xp(const float* __restrict__ seq, const float* __restrict__ g,
                                                      const float* __restrict__ bb, f16* __restrict__ xp) {
  __shared__ float red[256];
  long row = blockIdx.x;                 // 0..8*6085-1
  long b = row / 6085, n = row % 6085;
  const float* x = seq + row * 512;
  int t = threadIdx.x;
  float a0 = x[t], a1 = x[t + 256];
  red[t] = a0 + a1; __syncthreads();
  for (int s = 128; s > 0; s >>= 1) { if (t < s) red[t] += red[t + s]; __syncthreads(); }
  float mean = red[0] * (1.f / 512.f); __syncthreads();
  float d0 = a0 - mean, d1 = a1 - mean;
  red[t] = d0 * d0 + d1 * d1; __syncthreads();
  for (int s = 128; s > 0; s >>= 1) { if (t < s) red[t] += red[t + s]; __syncthreads(); }
  float inv = rsqrtf(red[0] * (1.f / 512.f) + 1e-5f);
  f16* o = xp + ((b * 6144) + 59 + n) * 512;
  o[t]       = (f16)(d0 * inv * g[t] + bb[t]);
  o[t + 256] = (f16)(d1 * inv * g[t + 256] + bb[t + 256]);
}

__global__ __launch_bounds__(256) void k_repack_qkv(const float* __restrict__ qkv, f16* __restrict__ q,
                                                    f16* __restrict__ k, f16* __restrict__ v) {
  long e = blockIdx.x * 256L + threadIdx.x;
  if (e >= 49152L * 1536) return;
  long rowi = e / 1536; int c = (int)(e % 1536);
  int b = (int)(rowi / 6144); int n = (int)(rowi % 6144);
  int which = c / 512; int cc = c % 512; int h = cc / 64; int d = cc % 64;
  float val = qkv[e];
  long dst = (((long)(b * 8 + h)) * 6144 + n) * 64 + d;
  if (which == 0)      q[dst] = (f16)(val * 0.125f);   // 1/sqrt(64)
  else if (which == 1) k[dst] = (f16)val;
  else                 v[dst] = (f16)val;
}

__global__ __launch_bounds__(256) void k_landmark(const f16* __restrict__ q, f16* __restrict__ ql) {
  long e = blockIdx.x * 256L + threadIdx.x;
  if (e >= 64L * 256 * 64) return;
  long bh = e >> 14; int rem = (int)(e & 16383); int m = rem >> 6; int d = rem & 63;
  const f16* src = q + (long)bh * 393216 + (long)m * 24 * 64 + d;
  float s = 0.f;
  for (int j = 0; j < 24; ++j) s += (float)src[(long)j * 64];
  ql[e] = (f16)(s * (1.f / 24.f));
}

__global__ __launch_bounds__(256) void k_softmax(const float* __restrict__ in, f16* __restrict__ out16,
                                                 float* __restrict__ out32, int len) {
  __shared__ float red[256];
  long row = blockIdx.x;
  const float* r = in + row * (long)len;
  int t = threadIdx.x;
  int per = len >> 8;            // 1 (len=256) or 24 (len=6144)
  float v[24];
  float mx = -3.4e38f;
  for (int i = 0; i < per; ++i) { v[i] = r[(long)i * 256 + t]; mx = fmaxf(mx, v[i]); }
  red[t] = mx; __syncthreads();
  for (int s = 128; s > 0; s >>= 1) { if (t < s) red[t] = fmaxf(red[t], red[t + s]); __syncthreads(); }
  mx = red[0]; __syncthreads();
  float sum = 0.f;
  for (int i = 0; i < per; ++i) { float e = __expf(v[i] - mx); v[i] = e; sum += e; }
  red[t] = sum; __syncthreads();
  for (int s = 128; s > 0; s >>= 1) { if (t < s) red[t] += red[t + s]; __syncthreads(); }
  float inv = 1.f / red[0];
  f16* o16 = out16 + row * (long)len;
  for (int i = 0; i < per; ++i) {
    float o = v[i] * inv;
    o16[(long)i * 256 + t] = (f16)o;
    if (out32) out32[row * (long)len + (long)i * 256 + t] = o;
  }
}

__global__ __launch_bounds__(256) void k_pinv_init(const float* __restrict__ x, f16* __restrict__ z) {
  __shared__ float red[256];
  long bh = blockIdx.x;
  const float* X = x + bh * 65536;
  int t = threadIdx.x;
  float rs = 0.f;
  for (int n = 0; n < 256; ++n) rs += fabsf(X[(long)t * 256 + n]);
  red[t] = rs; __syncthreads();
  for (int s = 128; s > 0; s >>= 1) { if (t < s) red[t] = fmaxf(red[t], red[t + s]); __syncthreads(); }
  float colv = red[0]; __syncthreads();
  float cs = 0.f;
  for (int m = 0; m < 256; ++m) cs += fabsf(X[(long)m * 256 + t]);
  red[t] = cs; __syncthreads();
  for (int s = 128; s > 0; s >>= 1) { if (t < s) red[t] = fmaxf(red[t], red[t + s]); __syncthreads(); }
  float rowv = red[0];
  float scale = 1.f / (colv * rowv);
  f16* Z = z + bh * 65536;
  for (int n = 0; n < 256; ++n) Z[(long)t * 256 + n] = (f16)(X[(long)n * 256 + t] * scale); // transpose
}

__global__ __launch_bounds__(256) void k_pinv_form(const float* __restrict__ in, f16* __restrict__ copy16,
                                                   f16* __restrict__ bout, float cdiag) {
  long e = blockIdx.x * 256L + threadIdx.x;
  if (e >= 64L * 65536) return;
  int loc = (int)(e & 65535); int i = loc >> 8; int j = loc & 255;
  float x = in[e];
  bout[e] = (f16)((i == j ? cdiag : 0.f) - x);
  if (copy16) copy16[e] = (f16)x;
}

__global__ __launch_bounds__(256) void k_resid_pack(const float* __restrict__ outAttn, const f16* __restrict__ v16,
                                                    const float* __restrict__ resw, const float* __restrict__ resb,
                                                    f16* __restrict__ cat) {
  long e = blockIdx.x * 256L + threadIdx.x;
  if (e >= 64L * 6144 * 64) return;
  long bh = e / 393216; int rem = (int)(e % 393216); int n = rem >> 6; int d = rem & 63;
  int b = (int)(bh >> 3); int h = (int)(bh & 7);
  float acc = outAttn[e] + resb[h];
  const f16* vv = v16 + bh * 393216 + d;
  const float* W = resw + h * 33;
  for (int j = 0; j < 33; ++j) {
    int nn = n + j - 16;
    if ((unsigned)nn < 6144u) acc += (float)vv[(long)nn * 64] * W[j];
  }
  cat[((long)(b * 6144 + n)) * 512 + h * 64 + d] = (f16)acc;
}

__global__ __launch_bounds__(256) void k_add_seq(float* __restrict__ seq, const float* __restrict__ P) {
  long e = blockIdx.x * 256L + threadIdx.x;
  if (e >= 8L * 6085 * 512) return;
  int d = (int)(e & 511);
  long rn = e >> 9;
  int n = (int)(rn % 6085);
  int b = (int)(rn / 6085);
  seq[e] += P[((long)b * 6144 + 59 + n) * 512 + d];
}

__global__ __launch_bounds__(256) void k_ppeg(const float* __restrict__ seq,
    const float* __restrict__ w7, const float* __restrict__ b7,
    const float* __restrict__ w5, const float* __restrict__ b5,
    const float* __restrict__ w3, const float* __restrict__ b3,
    float* __restrict__ tmp) {
  long e = blockIdx.x * 256L + threadIdx.x;
  if (e >= 8L * 6084 * 512) return;
  int d = (int)(e & 511);
  long nn = e >> 9;
  int n = (int)(nn % 6084);
  int b = (int)(nn / 6084);
  int r = n / 78, c = n % 78;
  const float* base = seq + ((long)b * 6085 + 1) * 512 + d;
  float acc = base[(long)n * 512] + b7[d] + b5[d] + b3[d];
  const float* W7 = w7 + d * 49;
  for (int kr = 0; kr < 7; ++kr) { int rr = r + kr - 3; if ((unsigned)rr < 78u)
    for (int kc = 0; kc < 7; ++kc) { int cc = c + kc - 3; if ((unsigned)cc < 78u)
      acc += base[((long)(rr * 78 + cc)) * 512] * W7[kr * 7 + kc]; } }
  const float* W5 = w5 + d * 25;
  for (int kr = 0; kr < 5; ++kr) { int rr = r + kr - 2; if ((unsigned)rr < 78u)
    for (int kc = 0; kc < 5; ++kc) { int cc = c + kc - 2; if ((unsigned)cc < 78u)
      acc += base[((long)(rr * 78 + cc)) * 512] * W5[kr * 5 + kc]; } }
  const float* W3 = w3 + d * 9;
  for (int kr = 0; kr < 3; ++kr) { int rr = r + kr - 1; if ((unsigned)rr < 78u)
    for (int kc = 0; kc < 3; ++kc) { int cc = c + kc - 1; if ((unsigned)cc < 78u)
      acc += base[((long)(rr * 78 + cc)) * 512] * W3[kr * 3 + kc]; } }
  tmp[e] = acc;
}

__global__ __launch_bounds__(256) void k_ppeg_copy(float* __restrict__ seq, const float* __restrict__ tmp) {
  long e = blockIdx.x * 256L + threadIdx.x;
  if (e >= 8L * 6084 * 512) return;
  int d = (int)(e & 511);
  long nn = e >> 9;
  int n = (int)(nn % 6084);
  int b = (int)(nn / 6084);
  seq[((long)b * 6085 + 1 + n) * 512 + d] = tmp[e];
}

__global__ __launch_bounds__(256) void k_ln_row0(const float* __restrict__ seq, const float* __restrict__ g,
                                                 const float* __restrict__ bb, float* __restrict__ bag) {
  __shared__ float red[256];
  int b = blockIdx.x;
  const float* x = seq + (long)b * 6085 * 512;
  int t = threadIdx.x;
  float a0 = x[t], a1 = x[t + 256];
  red[t] = a0 + a1; __syncthreads();
  for (int s = 128; s > 0; s >>= 1) { if (t < s) red[t] += red[t + s]; __syncthreads(); }
  float mean = red[0] * (1.f / 512.f); __syncthreads();
  float d0 = a0 - mean, d1 = a1 - mean;
  red[t] = d0 * d0 + d1 * d1; __syncthreads();
  for (int s = 128; s > 0; s >>= 1) { if (t < s) red[t] += red[t + s]; __syncthreads(); }
  float inv = rsqrtf(red[0] * (1.f / 512.f) + 1e-5f);
  bag[b * 512 + t]       = d0 * inv * g[t] + bb[t];
  bag[b * 512 + t + 256] = d1 * inv * g[t + 256] + bb[t + 256];
}

__global__ void k_mlp_head(const float* __restrict__ bag, const float* __restrict__ w,
                           const float* __restrict__ bb, float* __restrict__ out) {
  int t = threadIdx.x;
  if (t >= 16) return;
  int b = t >> 1, c = t & 1;
  float acc = bb[c];
  for (int d = 0; d < 512; ++d) acc += bag[b * 512 + d] * w[d * 2 + c];
  out[t] = acc;
}

__global__ __launch_bounds__(256) void k_build_xcat(const float* __restrict__ bag, const float* __restrict__ reh,
                                                    float* __restrict__ xcat) {
  long e = blockIdx.x * 256L + threadIdx.x;
  if (e >= 520L * 512) return;
  int i = (int)(e >> 9); int d = (int)(e & 511);
  xcat[e] = (i < 8) ? bag[i * 512 + d] : reh[(long)(i - 8) * 512 + d];
}

__global__ __launch_bounds__(256) void k_rownorm(const float* __restrict__ xcat, float* __restrict__ xn) {
  __shared__ float red[256];
  int row = blockIdx.x;
  const float* x = xcat + (long)row * 512;
  int t = threadIdx.x;
  float a0 = x[t], a1 = x[t + 256];
  red[t] = a0 * a0 + a1 * a1; __syncthreads();
  for (int s = 128; s > 0; s >>= 1) { if (t < s) red[t] += red[t + s]; __syncthreads(); }
  float inv = 1.f / (sqrtf(red[0]) + 1e-8f);
  xn[(long)row * 512 + t]       = a0 * inv;
  xn[(long)row * 512 + t + 256] = a1 * inv;
}

__global__ __launch_bounds__(256) void k_sim(const float* __restrict__ xn, float* __restrict__ sim) {
  long e = blockIdx.x * 256L + threadIdx.x;
  if (e >= 520L * 520) return;
  int i = (int)(e / 520); int j = (int)(e % 520);
  const float* a = xn + (long)i * 512;
  const float* b = xn + (long)j * 512;
  float s = 0.f;
  for (int d = 0; d < 512; ++d) s += a[d] * b[d];
  if (i == j) s -= 1e9f;
  sim[e] = s;
}

__global__ __launch_bounds__(256) void k_topk(const float* __restrict__ sim, float* __restrict__ vals,
                                              int* __restrict__ idx) {
  int i = blockIdx.x * 256 + threadIdx.x;
  if (i >= 520) return;
  const float* r = sim + (long)i * 520;
  float pv = 3.4e38f; int pi = -1;
  for (int p = 0; p < 12; ++p) {
    float best = -3.4e38f; int bi = 0;
    for (int j = 0; j < 520; ++j) {
      float s = r[j];
      bool elig = (s < pv) || (s == pv && j > pi);
      if (elig && s > best) { best = s; bi = j; }
    }
    vals[i * 12 + p] = best; idx[i * 12 + p] = bi; pv = best; pi = bi;
  }
}

__global__ __launch_bounds__(256) void k_deg(const float* __restrict__ vals, float* __restrict__ dinv) {
  int i = blockIdx.x * 256 + threadIdx.x;
  if (i >= 520) return;
  float d = 1.f;
  for (int k = 0; k < 12; ++k) { float w = vals[i * 12 + k]; d += (w > 0.f ? w : 0.f); }
  dinv[i] = rsqrtf(d);
}

__global__ __launch_bounds__(256) void k_pf(const float* __restrict__ xcat, float* __restrict__ pf) {
  long e = blockIdx.x * 256L + threadIdx.x;
  if (e >= 520L * 512) return;
  int i = (int)(e >> 9);
  pf[e] = (i < 8) ? xcat[e] : 0.f;
}

__global__ __launch_bounds__(256) void k_mm_small(const float* __restrict__ feat, const float* __restrict__ w,
                                                  float* __restrict__ out, int rows, int fin, int fout) {
  long e = blockIdx.x * 256L + threadIdx.x;
  if (e >= (long)rows * fout) return;
  int i = (int)(e / fout); int o = (int)(e % fout);
  float acc = 0.f;
  const float* f = feat + (long)i * fin;
  for (int k = 0; k < fin; ++k) acc += f[k] * w[(long)k * fout + o];
  out[e] = acc;
}

__global__ __launch_bounds__(256) void k_gcn_combine(const float* __restrict__ hh, const float* __restrict__ vals,
                                                     const int* __restrict__ idx, const float* __restrict__ dinv,
                                                     const float* __restrict__ bias, float* __restrict__ out,
                                                     int fout, int doRelu) {
  long e = blockIdx.x * 256L + threadIdx.x;
  if (e >= 520L * fout) return;
  int i = (int)(e / fout); int o = (int)(e % fout);
  float di = dinv[i];
  float acc = hh[(long)i * fout + o] * di * di + bias[o];
  for (int k = 0; k < 12; ++k) {
    float w = vals[i * 12 + k]; w = w > 0.f ? w : 0.f;
    int s = idx[i * 12 + k];
    acc += w * dinv[s] * di * hh[(long)s * fout + o];
  }
  if (doRelu) acc = fmaxf(acc, 0.f);
  out[e] = acc;
}

__global__ void k_copy_out(const float* __restrict__ g2, float* __restrict__ out) {
  int t = threadIdx.x;
  if (t < 16) out[t] = g2[t];   // rows 0..7 of (520,2) are first 16 elems
}

// ======================= host orchestration =======================
extern "C" void kernel_launch(void* const* d_in, const int* in_sizes, int n_in,
                              void* d_out, int out_size, void* d_ws, size_t ws_size,
                              hipStream_t stream) {
  (void)in_sizes; (void)n_in; (void)out_size; (void)ws_size;
  const float* x     = (const float*)d_in[0];
  const float* fc1_w = (const float*)d_in[1];
  const float* fc1_b = (const float*)d_in[2];
  const float* cls   = (const float*)d_in[3];
  const float* lng[2]  = {(const float*)d_in[4],  (const float*)d_in[11]};
  const float* lnb[2]  = {(const float*)d_in[5],  (const float*)d_in[12]};
  const float* qkvw[2] = {(const float*)d_in[6],  (const float*)d_in[13]};
  const float* outw[2] = {(const float*)d_in[7],  (const float*)d_in[14]};
  const float* outb[2] = {(const float*)d_in[8],  (const float*)d_in[15]};
  const float* resw[2] = {(const float*)d_in[9],  (const float*)d_in[16]};
  const float* resb[2] = {(const float*)d_in[10], (const float*)d_in[17]};
  const float* w7 = (const float*)d_in[18]; const float* b7 = (const float*)d_in[19];
  const float* w5 = (const float*)d_in[20]; const float* b5 = (const float*)d_in[21];
  const float* w3 = (const float*)d_in[22]; const float* b3 = (const float*)d_in[23];
  const float* lnfg = (const float*)d_in[24]; const float* lnfb = (const float*)d_in[25];
  const float* fc2w = (const float*)d_in[26]; const float* fc2b = (const float*)d_in[27];
  const float* gw1 = (const float*)d_in[28]; const float* gb1 = (const float*)d_in[29];
  const float* gw2 = (const float*)d_in[30]; const float* gb2 = (const float*)d_in[31];
  const float* reh = (const float*)d_in[32];
  float* out = (float*)d_out;

  char* ws = (char*)d_ws;
  size_t off = 0;
  auto alloc = [&](size_t bytes) -> char* {
    char* p = ws + off;
    off += (bytes + 255) & ~(size_t)255;
    return p;
  };
  f16* fc1w16    = (f16*)alloc(262144L * 2);
  f16* qkvw16_0  = (f16*)alloc(786432L * 2);
  f16* qkvw16_1  = (f16*)alloc(786432L * 2);
  f16* outw16_0  = (f16*)alloc(262144L * 2);
  f16* outw16_1  = (f16*)alloc(262144L * 2);
  f16* catArena  = (f16*)alloc(25165824L * 2);   // x16, then out-attn cat per layer
  float* seq     = (float*)alloc(24924160L * 4); // (8,6085,512)
  f16* xp16      = (f16*)alloc(25165824L * 2);   // (8,6144,512) padded LN output
  float* big1    = (float*)alloc(25165824L * 4); // h0 / out_attn / out-proj / ppeg tmp
  float* logits  = (float*)alloc(100663296L * 4);// qkv32 / attn3 logits / attn1 logits
  f16* att16     = (f16*)alloc(100663296L * 2);  // attn3 f16 then attn1 f16
  f16* q16       = (f16*)alloc(25165824L * 2);
  f16* k16       = (f16*)alloc(25165824L * 2);
  f16* v16       = (f16*)alloc(25165824L * 2);
  f16* ql16      = (f16*)alloc(1048576L * 2);
  f16* kl16      = (f16*)alloc(1048576L * 2);
  float* a2_32   = (float*)alloc(4194304L * 4);
  f16* a2_16     = (f16*)alloc(4194304L * 2);
  float* z32     = (float*)alloc(4194304L * 4);
  f16* z16       = (f16*)alloc(4194304L * 2);
  float* xz32    = (float*)alloc(4194304L * 4);
  f16* xz16      = (f16*)alloc(4194304L * 2);
  f16* bmat16    = (f16*)alloc(4194304L * 2);
  float* cbuf    = (float*)alloc(4194304L * 4);
  float* t32     = (float*)alloc(1048576L * 4);
  f16* t16       = (f16*)alloc(1048576L * 2);
  float* u32     = (float*)alloc(1048576L * 4);
  f16* u16       = (f16*)alloc(1048576L * 2);
  float* bag     = (float*)alloc(4096L * 4);
  float* xcat    = (float*)alloc(266240L * 4);
  float* xn      = (float*)alloc(266240L * 4);
  float* pf      = (float*)alloc(266240L * 4);
  float* sim     = (float*)alloc(270400L * 4);
  float* vals    = (float*)alloc(6240L * 4);
  int*   idxb    = (int*)alloc(6240L * 4);
  float* dinv    = (float*)alloc(520L * 4);
  float* hh1     = (float*)alloc(133120L * 4);
  float* g1      = (float*)alloc(133120L * 4);
  float* hh2     = (float*)alloc(1040L * 4);
  float* g2      = (float*)alloc(1040L * 4);
  f16* qkvw16[2] = {qkvw16_0, qkvw16_1};
  f16* outw16[2] = {outw16_0, outw16_1};

  auto EW = [](long n) { return dim3((unsigned)((n + 255) / 256)); };
  auto gemm = [&](const f16* A, const f16* B, float* C, const float* bias,
                  int M, int N, int K, long lda, long ldb, long ldc,
                  long sA, long sB, long sC, int batch, int opB, int epi, float alpha) {
    k_gemm<<<dim3((unsigned)(N / 64), (unsigned)(M / 128), (unsigned)batch), 256, 0, stream>>>(
        A, B, C, bias, M, N, K, lda, ldb, ldc, sA, sB, sC, opB, epi, alpha);
  };

  // ---- weight + input conversion ----
  k_cast_f16<<<EW(262144), 256, 0, stream>>>(fc1_w, fc1w16, 262144);
  k_cast_f16<<<EW(786432), 256, 0, stream>>>(qkvw[0], qkvw16[0], 786432);
  k_cast_f16<<<EW(786432), 256, 0, stream>>>(qkvw[1], qkvw16[1], 786432);
  k_cast_f16<<<EW(262144), 256, 0, stream>>>(outw[0], outw16[0], 262144);
  k_cast_f16<<<EW(262144), 256, 0, stream>>>(outw[1], outw16[1], 262144);
  k_cast_f16<<<EW(24576000L), 256, 0, stream>>>(x, catArena, 24576000L);

  // ---- fc1 + relu, build sequence with cls token + wraparound ----
  gemm(catArena, fc1w16, big1, fc1_b, 48000, 512, 512, 512, 512, 512, 0, 0, 0, 1, 0, 2, 1.f);
  k_build_seq<<<EW(24924160L), 256, 0, stream>>>(big1, cls, seq);
  k_zero_pad<<<EW(241664L), 256, 0, stream>>>(xp16);

  for (int l = 0; l < 2; ++l) {
    // LN -> padded f16
    k_layernorm_xp<<<48680, 256, 0, stream>>>(seq, lng[l], lnb[l], xp16);
    // QKV projection
    gemm(xp16, qkvw16[l], logits, nullptr, 49152, 1536, 512, 512, 1536, 1536, 0, 0, 0, 1, 0, 0, 1.f);
    k_repack_qkv<<<EW(75497472L), 256, 0, stream>>>(logits, q16, k16, v16);
    k_landmark<<<EW(1048576L), 256, 0, stream>>>(q16, ql16);
    k_landmark<<<EW(1048576L), 256, 0, stream>>>(k16, kl16);
    // attn2 = softmax(q_l @ k_l^T), then Newton-Schulz pseudo-inverse
    gemm(ql16, kl16, a2_32, nullptr, 256, 256, 64, 64, 64, 256, 16384, 16384, 65536, 64, 1, 0, 1.f);
    k_softmax<<<16384, 256, 0, stream>>>(a2_32, a2_16, a2_32, 256);
    k_pinv_init<<<64, 256, 0, stream>>>(a2_32, z16);
    for (int it = 0; it < 6; ++it) {
      gemm(a2_16, z16, xz32, nullptr, 256, 256, 256, 256, 256, 256, 65536, 65536, 65536, 64, 0, 0, 1.f);
      k_pinv_form<<<EW(4194304L), 256, 0, stream>>>(xz32, xz16, bmat16, 7.f);
      gemm(xz16, bmat16, cbuf, nullptr, 256, 256, 256, 256, 256, 256, 65536, 65536, 65536, 64, 0, 0, 1.f);
      k_pinv_form<<<EW(4194304L), 256, 0, stream>>>(cbuf, (f16*)nullptr, bmat16, 15.f);
      gemm(xz16, bmat16, cbuf, nullptr, 256, 256, 256, 256, 256, 256, 65536, 65536, 65536, 64, 0, 0, 1.f);
      k_pinv_form<<<EW(4194304L), 256, 0, stream>>>(cbuf, (f16*)nullptr, bmat16, 13.f);
      gemm(z16, bmat16, z32, nullptr, 256, 256, 256, 256, 256, 256, 65536, 65536, 65536, 64, 0, 0, 0.25f);
      k_cast_f16<<<EW(4194304L), 256, 0, stream>>>(z32, z16, 4194304L);
    }
    // attn3 = softmax(q_l @ k^T); t = attn3 @ v; u = Z @ t
    gemm(ql16, k16, logits, nullptr, 256, 6144, 64, 64, 64, 6144, 16384, 393216, 1572864, 64, 1, 0, 1.f);
    k_softmax<<<16384, 256, 0, stream>>>(logits, att16, nullptr, 6144);
    gemm(att16, v16, t32, nullptr, 256, 64, 6144, 6144, 64, 64, 1572864, 393216, 16384, 64, 0, 0, 1.f);
    k_cast_f16<<<EW(1048576L), 256, 0, stream>>>(t32, t16, 1048576L);
    gemm(z16, t16, u32, nullptr, 256, 64, 256, 256, 64, 64, 65536, 16384, 16384, 64, 0, 0, 1.f);
    k_cast_f16<<<EW(1048576L), 256, 0, stream>>>(u32, u16, 1048576L);
    // attn1 = softmax(q @ k_l^T); out = attn1 @ u
    gemm(q16, kl16, logits, nullptr, 6144, 256, 64, 64, 64, 256, 393216, 16384, 1572864, 64, 1, 0, 1.f);
    k_softmax<<<393216, 256, 0, stream>>>(logits, att16, nullptr, 256);
    gemm(att16, u16, big1, nullptr, 6144, 64, 256, 256, 64, 64, 1572864, 16384, 393216, 64, 0, 0, 1.f);
    // + depthwise residual conv + bias, transpose-pack to (B*N, 512) f16
    k_resid_pack<<<EW(25165824L), 256, 0, stream>>>(big1, v16, resw[l], resb[l], catArena);
    // output projection + bias; residual add into sequence
    gemm(catArena, outw16[l], big1, outb[l], 49152, 512, 512, 512, 512, 512, 0, 0, 0, 1, 0, 1, 1.f);
    k_add_seq<<<EW(24924160L), 256, 0, stream>>>(seq, big1);
    if (l == 0) {
      k_ppeg<<<EW(24920064L), 256, 0, stream>>>(seq, w7, b7, w5, b5, w3, b3, big1);
      k_ppeg_copy<<<EW(24920064L), 256, 0, stream>>>(seq, big1);
    }
  }

  // ---- heads ----
  k_ln_row0<<<8, 256, 0, stream>>>(seq, lnfg, lnfb, bag);
  k_mlp_head<<<1, 16, 0, stream>>>(bag, fc2w, fc2b, out);
  k_build_xcat<<<EW(266240L), 256, 0, stream>>>(bag, reh, xcat);
  k_rownorm<<<520, 256, 0, stream>>>(xcat, xn);
  k_sim<<<EW(270400L), 256, 0, stream>>>(xn, sim);
  k_topk<<<EW(520L), 256, 0, stream>>>(sim, vals, idxb);
  k_deg<<<EW(520L), 256, 0, stream>>>(vals, dinv);
  k_pf<<<EW(266240L), 256, 0, stream>>>(xcat, pf);
  k_mm_small<<<EW(133120L), 256, 0, stream>>>(pf, gw1, hh1, 520, 512, 256);
  k_gcn_combine<<<EW(133120L), 256, 0, stream>>>(hh1, vals, idxb, dinv, gb1, g1, 256, 1);
  k_mm_small<<<EW(1040L), 256, 0, stream>>>(g1, gw2, hh2, 520, 256, 2);
  k_gcn_combine<<<EW(1040L), 256, 0, stream>>>(hh2, vals, idxb, dinv, gb2, g2, 2, 0);
  k_copy_out<<<1, 16, 0, stream>>>(g2, out + 16);
}
